// LlamaDecoderLayer_86440511800077
// MI455X (gfx1250) — compile-verified, matrix operand order checked by reading, not run on hardware
//
#include <hip/hip_runtime.h>
#include <math.h>

typedef __bf16 bf16;
typedef __attribute__((ext_vector_type(8)))  bf16  bf16x8;
typedef __attribute__((ext_vector_type(16))) bf16  bf16x16;
typedef __attribute__((ext_vector_type(8)))  float v8f;
typedef __attribute__((ext_vector_type(4)))  int   i32x4;

#define B_    4
#define S_    1024
#define D_    1024
#define H_    16
#define QL_   512
#define KVL_  256
#define NOPE_ 64
#define ROPE_ 32
#define VD_   64
#define E_    8
#define FE_   1024
#define TOK_  (B_*S_)
#define NEGF  (-1.0e15f)

union FragU { bf16x16 v; bf16x8 h[2]; };

// ---------------------------------------------------------------------------
// Async global->LDS copy path (CDNA5); falls back to load+store if the
// builtin is unavailable on this toolchain.
// Probe-derived signature: (v4i addrspace(1)* src, v4i addrspace(3)* dst,
//                           imm offset, imm cpol)
// ---------------------------------------------------------------------------
#if __has_builtin(__builtin_amdgcn_global_load_async_to_lds_b128)
#define ASYNC_MODE 1
#endif

typedef __attribute__((address_space(1))) i32x4 as1_i32x4;
typedef __attribute__((address_space(3))) i32x4 as3_i32x4;

__device__ __forceinline__ void cp16(bf16* l, const bf16* g) {
#ifdef ASYNC_MODE
  __builtin_amdgcn_global_load_async_to_lds_b128((as1_i32x4*)g, (as3_i32x4*)l, 0, 0);
#else
  *(bf16x8*)l = *(const bf16x8*)g;
#endif
}

__device__ __forceinline__ void async_fence() {
#ifdef ASYNC_MODE
#if __has_builtin(__builtin_amdgcn_s_wait_asynccnt)
  __builtin_amdgcn_s_wait_asynccnt(0);
#else
  asm volatile("s_wait_asynccnt 0" ::: "memory");
#endif
#endif
}

// ---------------------------------------------------------------------------
// Generic bf16 WMMA GEMM:  C[M,N] = scale * (A[M,K] @ W[N,K]^T) + bias + add
// Block: 128 threads = 4 waves; 64x64 output tile.
// K staged via double-buffered LDS in 64-wide slabs (async copy -> LDS),
// one barrier per slab, 8 WMMAs per wave per slab.
// Batched over gridDim.z with z = b*hdiv + h and per-(b,h) strides.
// Requirements: M % 64 == 0, K % 32 == 0 (N may be ragged).
// ---------------------------------------------------------------------------
template<bool OUT_BF16>
__global__ __launch_bounds__(128)
void k_gemm(const bf16* __restrict__ Ag, const bf16* __restrict__ Wg,
            const float* __restrict__ bias, const float* __restrict__ addsrc,
            void* __restrict__ Cout,
            int M, int N, int K, int lda, int ldw, int ldc,
            long long aB, long long aH, long long wB, long long wH,
            long long cB, long long cH, float scale,
            int causalScore, int causalK, int hdiv)
{
  constexpr int LD = 72; // padded LDS row stride (elems); 144B keeps 16B align
  __shared__ bf16 As[2][64 * LD];
  __shared__ bf16 Bs[2][64 * LD];

  const int z  = blockIdx.z;
  const int bi = z / hdiv, hi = z % hdiv;
  Ag += bi * aB + hi * aH;
  Wg += bi * wB + hi * wH;
  const long long cOff = bi * cB + hi * cH;

  const int rowBase = blockIdx.y * 64;
  const int colBase = blockIdx.x * 64;
  if (causalScore && colBase > rowBase + 63) return;   // fully-masked tile
  int kEnd = K;
  if (causalK) { int lim = rowBase + 64; kEnd = (lim < K) ? lim : K; }

  const int tid  = threadIdx.x;
  const int lrow = tid >> 1;          // 0..63 tile row loaded by this thread
  const int kcol = (tid & 1) << 5;    // 0 or 32 (k offset inside slab)
  const int lane = tid & 31;
  const int wv   = tid >> 5;
  const int wr   = (wv >> 1) * 32;    // wave row offset in tile
  const int wc   = (wv & 1) * 32;     // wave col offset in tile
  const int half = lane >> 4;
  const int l16  = lane & 15;

  v8f acc[2][2];
#pragma unroll
  for (int i = 0; i < 2; ++i)
#pragma unroll
    for (int j = 0; j < 2; ++j)
#pragma unroll
      for (int r = 0; r < 8; ++r) acc[i][j][r] = 0.0f;

  bf16x8 zv;
#pragma unroll
  for (int i = 0; i < 8; ++i) zv[i] = (bf16)0.0f;

  // copy one 64-wide K slab (this thread's 64B strip of A and W) into LDS
  auto copySlab = [&](int k0, bf16* Asb, bf16* Bsb) {
    bool kv = (kcol == 0) || (k0 + 32 < kEnd);   // 32-granular K remainder
    if (!kv) return;
    {
      const bf16* g = Ag + (size_t)(rowBase + lrow) * lda + k0 + kcol;
      bf16* l = Asb + lrow * LD + kcol;
      cp16(l, g); cp16(l + 8, g + 8); cp16(l + 16, g + 16); cp16(l + 24, g + 24);
      if (k0 + 64 < kEnd) __builtin_prefetch(g + 64, 0, 0);
    }
    {
      int wrr = colBase + lrow;
      bf16* l = Bsb + lrow * LD + kcol;
      if (wrr < N) {
        const bf16* g = Wg + (size_t)wrr * ldw + k0 + kcol;
        cp16(l, g); cp16(l + 8, g + 8); cp16(l + 16, g + 16); cp16(l + 24, g + 24);
        if (k0 + 64 < kEnd) __builtin_prefetch(g + 64, 0, 0);
      } else {
        *(bf16x8*)l = zv; *(bf16x8*)(l + 8) = zv;
        *(bf16x8*)(l + 16) = zv; *(bf16x8*)(l + 24) = zv;
      }
    }
  };

  // one 32-deep WMMA step on the current slab (ks = 0 or 32 inside slab)
  auto mmStep = [&](const bf16* Asb, const bf16* Bsb, int ks) {
    FragU fa[2], fb[2];
#pragma unroll
    for (int i = 0; i < 2; ++i) {
      const bf16* ap = Asb + (wr + i * 16 + l16) * LD + ks;
      fa[i].h[0] = *(const bf16x8*)(ap + (half ? 8 : 0));
      fa[i].h[1] = *(const bf16x8*)(ap + (half ? 24 : 16));
      const bf16* bp = Bsb + (wc + i * 16 + l16) * LD + ks;
      fb[i].h[0] = *(const bf16x8*)(bp + (half ? 8 : 0));
      fb[i].h[1] = *(const bf16x8*)(bp + (half ? 24 : 16));
    }
#pragma unroll
    for (int i = 0; i < 2; ++i)
#pragma unroll
      for (int j = 0; j < 2; ++j)
        acc[i][j] = __builtin_amdgcn_wmma_f32_16x16x32_bf16(
            false, fa[i].v, false, fb[j].v, (short)0, acc[i][j], false, false);
  };

  // software pipeline: prologue copy, then (wait; copy next; compute current)
  copySlab(0, As[0], Bs[0]);
  int cur = 0;
  for (int k0 = 0; k0 < kEnd; k0 += 64, cur ^= 1) {
    async_fence();
    __syncthreads();   // slab `cur` resident; all waves done reading `cur^1`
    if (k0 + 64 < kEnd) copySlab(k0 + 64, As[cur ^ 1], Bs[cur ^ 1]);
    mmStep(As[cur], Bs[cur], 0);
    if (k0 + 32 < kEnd) mmStep(As[cur], Bs[cur], 32);
  }

#pragma unroll
  for (int i = 0; i < 2; ++i) {
#pragma unroll
    for (int j = 0; j < 2; ++j) {
      int ccol = colBase + wc + j * 16 + l16;
      if (ccol >= N) continue;
      float bv = bias ? bias[ccol] : 0.0f;
      int rb = rowBase + wr + i * 16 + half * 8;
#pragma unroll
      for (int r = 0; r < 8; ++r) {
        int row = rb + r;
        if (row >= M) continue;
        size_t idx = (size_t)cOff + (size_t)row * ldc + ccol;
        float v = acc[i][j][r] * scale + bv;
        if (addsrc) v += addsrc[idx];
        if (OUT_BF16) ((bf16*)Cout)[idx] = (bf16)v;
        else          ((float*)Cout)[idx] = v;
      }
    }
  }
}

// ---------------------------------------------------------------------------
// MoE fused gate/up GEMM + SiLU  (gathered rows for one expert per block.z)
// ---------------------------------------------------------------------------
__global__ __launch_bounds__(128)
void k_moe_gateup(const bf16* __restrict__ Yb, const bf16* __restrict__ Gw,
                  const bf16* __restrict__ Uw, const int* __restrict__ counts,
                  const int* __restrict__ rowsIdx, bf16* __restrict__ Hact)
{
  constexpr int LD = 40;
  __shared__ bf16 As[64 * LD];
  __shared__ bf16 Gs[64 * LD];
  __shared__ bf16 Us[64 * LD];

  const int e  = blockIdx.z;
  const int ne = counts[e];
  const int rowBase = blockIdx.y * 64;
  if (rowBase >= ne) return;
  const int colBase = blockIdx.x * 64;
  const int* rowsE = rowsIdx + e * TOK_;
  const bf16* Ge = Gw + (size_t)e * FE_ * D_;
  const bf16* Ue = Uw + (size_t)e * FE_ * D_;

  const int tid  = threadIdx.x;
  const int lr   = tid >> 1;
  const int lc   = (tid & 1) << 4;
  const int lane = tid & 31;
  const int wv   = tid >> 5;
  const int wr   = (wv >> 1) * 32;
  const int wc   = (wv & 1) * 32;
  const int half = lane >> 4;
  const int l16  = lane & 15;

  int grow = rowBase + lr;
  int tok = (grow < ne) ? rowsE[grow] : -1;

  v8f aG[2][2], aU[2][2];
#pragma unroll
  for (int i = 0; i < 2; ++i)
#pragma unroll
    for (int j = 0; j < 2; ++j)
#pragma unroll
      for (int r = 0; r < 8; ++r) { aG[i][j][r] = 0.0f; aU[i][j][r] = 0.0f; }

  bf16x8 zv;
#pragma unroll
  for (int i = 0; i < 8; ++i) zv[i] = (bf16)0.0f;

  for (int k0 = 0; k0 < D_; k0 += 32) {
    __syncthreads();   // previous iteration's MMAs done before overwrite
    if (tok >= 0) {
      const bf16* p = Yb + (size_t)tok * D_ + k0 + lc;
      bf16* l = &As[lr * LD + lc];
      cp16(l, p); cp16(l + 8, p + 8);
    } else {
      *(bf16x8*)&As[lr * LD + lc] = zv;
      *(bf16x8*)&As[lr * LD + lc + 8] = zv;
    }
    {
      const bf16* pg = Ge + (size_t)(colBase + lr) * D_ + k0 + lc;
      bf16* l = &Gs[lr * LD + lc];
      cp16(l, pg); cp16(l + 8, pg + 8);
      const bf16* pu = Ue + (size_t)(colBase + lr) * D_ + k0 + lc;
      bf16* lu = &Us[lr * LD + lc];
      cp16(lu, pu); cp16(lu + 8, pu + 8);
    }
    async_fence();
    __syncthreads();

#pragma unroll
    for (int i = 0; i < 2; ++i) {
      FragU fa;
      const bf16* ap = &As[(wr + i * 16 + l16) * LD];
      fa.h[0] = *(const bf16x8*)(ap + (half ? 8 : 0));
      fa.h[1] = *(const bf16x8*)(ap + (half ? 24 : 16));
#pragma unroll
      for (int j = 0; j < 2; ++j) {
        FragU fg, fu;
        const bf16* gp = &Gs[(wc + j * 16 + l16) * LD];
        fg.h[0] = *(const bf16x8*)(gp + (half ? 8 : 0));
        fg.h[1] = *(const bf16x8*)(gp + (half ? 24 : 16));
        const bf16* up = &Us[(wc + j * 16 + l16) * LD];
        fu.h[0] = *(const bf16x8*)(up + (half ? 8 : 0));
        fu.h[1] = *(const bf16x8*)(up + (half ? 24 : 16));
        aG[i][j] = __builtin_amdgcn_wmma_f32_16x16x32_bf16(
            false, fa.v, false, fg.v, (short)0, aG[i][j], false, false);
        aU[i][j] = __builtin_amdgcn_wmma_f32_16x16x32_bf16(
            false, fa.v, false, fu.v, (short)0, aU[i][j], false, false);
      }
    }
  }

#pragma unroll
  for (int i = 0; i < 2; ++i) {
#pragma unroll
    for (int j = 0; j < 2; ++j) {
      int ccol = colBase + wc + j * 16 + l16;
      int rb = rowBase + wr + i * 16 + half * 8;
#pragma unroll
      for (int r = 0; r < 8; ++r) {
        int row = rb + r;
        if (row < ne) {
          float g = aG[i][j][r];
          float u = aU[i][j][r];
          float hval = (g / (1.0f + expf(-g))) * u;  // silu(g)*u
          Hact[((size_t)e * TOK_ + row) * FE_ + ccol] = (bf16)hval;
        }
      }
    }
  }
}

// ---------------------------------------------------------------------------
// MoE down-proj GEMM with weighted scatter-add into the output residual
// ---------------------------------------------------------------------------
__global__ __launch_bounds__(128)
void k_moe_down(const bf16* __restrict__ Hact, const bf16* __restrict__ Dw,
                const int* __restrict__ counts, const int* __restrict__ rowsIdx,
                const float* __restrict__ wts, float* __restrict__ Out)
{
  constexpr int LD = 40;
  __shared__ bf16 As[64 * LD];
  __shared__ bf16 Bs[64 * LD];

  const int e  = blockIdx.z;
  const int ne = counts[e];
  const int rowBase = blockIdx.y * 64;
  if (rowBase >= ne) return;
  const int colBase = blockIdx.x * 64;
  const int* rowsE = rowsIdx + e * TOK_;
  const float* wtsE = wts + e * TOK_;
  const bf16* De = Dw + (size_t)e * D_ * FE_;

  const int tid  = threadIdx.x;
  const int lr   = tid >> 1;
  const int lc   = (tid & 1) << 4;
  const int lane = tid & 31;
  const int wv   = tid >> 5;
  const int wr   = (wv >> 1) * 32;
  const int wc   = (wv & 1) * 32;
  const int half = lane >> 4;
  const int l16  = lane & 15;

  v8f acc[2][2];
#pragma unroll
  for (int i = 0; i < 2; ++i)
#pragma unroll
    for (int j = 0; j < 2; ++j)
#pragma unroll
      for (int r = 0; r < 8; ++r) acc[i][j][r] = 0.0f;

  bf16x8 zv;
#pragma unroll
  for (int i = 0; i < 8; ++i) zv[i] = (bf16)0.0f;

  int grow = rowBase + lr;

  for (int k0 = 0; k0 < FE_; k0 += 32) {
    __syncthreads();
    if (grow < ne) {
      const bf16* p = Hact + ((size_t)e * TOK_ + grow) * FE_ + k0 + lc;
      bf16* l = &As[lr * LD + lc];
      cp16(l, p); cp16(l + 8, p + 8);
    } else {
      *(bf16x8*)&As[lr * LD + lc] = zv;
      *(bf16x8*)&As[lr * LD + lc + 8] = zv;
    }
    {
      const bf16* pw = De + (size_t)(colBase + lr) * FE_ + k0 + lc;
      bf16* l = &Bs[lr * LD + lc];
      cp16(l, pw); cp16(l + 8, pw + 8);
    }
    async_fence();
    __syncthreads();

#pragma unroll
    for (int i = 0; i < 2; ++i) {
      FragU fa;
      const bf16* ap = &As[(wr + i * 16 + l16) * LD];
      fa.h[0] = *(const bf16x8*)(ap + (half ? 8 : 0));
      fa.h[1] = *(const bf16x8*)(ap + (half ? 24 : 16));
#pragma unroll
      for (int j = 0; j < 2; ++j) {
        FragU fb;
        const bf16* bp = &Bs[(wc + j * 16 + l16) * LD];
        fb.h[0] = *(const bf16x8*)(bp + (half ? 8 : 0));
        fb.h[1] = *(const bf16x8*)(bp + (half ? 24 : 16));
        acc[i][j] = __builtin_amdgcn_wmma_f32_16x16x32_bf16(
            false, fa.v, false, fb.v, (short)0, acc[i][j], false, false);
      }
    }
  }

#pragma unroll
  for (int i = 0; i < 2; ++i) {
#pragma unroll
    for (int j = 0; j < 2; ++j) {
      int ccol = colBase + wc + j * 16 + l16;
      int rb = rowBase + wr + i * 16 + half * 8;
#pragma unroll
      for (int r = 0; r < 8; ++r) {
        int row = rb + r;
        if (row < ne) {
          int   tok = rowsE[row];
          float wgt = wtsE[row];
          atomicAdd(&Out[(size_t)tok * D_ + ccol], acc[i][j][r] * wgt);
        }
      }
    }
  }
}

// ---------------------------------------------------------------------------
// Elementwise / small kernels
// ---------------------------------------------------------------------------
__global__ void k_cast_bf16(const float* __restrict__ s, bf16* __restrict__ d, size_t n) {
  size_t i = (size_t)blockIdx.x * blockDim.x + threadIdx.x;
  size_t st = (size_t)gridDim.x * blockDim.x;
  for (; i < n; i += st) d[i] = (bf16)s[i];
}

__global__ void k_copy_f32(const float* __restrict__ s, float* __restrict__ d, size_t n) {
  size_t i = (size_t)blockIdx.x * blockDim.x + threadIdx.x;
  size_t st = (size_t)gridDim.x * blockDim.x;
  for (; i < n; i += st) d[i] = s[i];
}

__global__ void k_zero_i32(int* __restrict__ p, int n) {
  int i = blockIdx.x * blockDim.x + threadIdx.x;
  if (i < n) p[i] = 0;
}

// rms_norm over one row per block, bf16 output
__global__ __launch_bounds__(256)
void k_rmsnorm(const float* __restrict__ X, const float* __restrict__ w,
               bf16* __restrict__ Y, int dim)
{
  __shared__ float red[256];
  int row = blockIdx.x;
  const float* x = X + (size_t)row * dim;
  float ss = 0.0f;
  for (int i = threadIdx.x; i < dim; i += 256) { float v = x[i]; ss += v * v; }
  red[threadIdx.x] = ss;
  __syncthreads();
  for (int s = 128; s > 0; s >>= 1) {
    if (threadIdx.x < s) red[threadIdx.x] += red[threadIdx.x + s];
    __syncthreads();
  }
  float r = rsqrtf(red[0] / (float)dim + 1e-6f);
  for (int i = threadIdx.x; i < dim; i += 256)
    Y[(size_t)row * dim + i] = (bf16)(w[i] * x[i] * r);
}

// split wkv_b (H*(NOPE+VD), KVL) into nope^T (H,256,64) and v (H,64,256), bf16
__global__ void k_wkvb_split(const float* __restrict__ w,
                             bf16* __restrict__ nopeT, bf16* __restrict__ vmat)
{
  int i = blockIdx.x * 256 + threadIdx.x;     // grid 2048 x 256
  int row = i >> 8, c = i & 255;
  int h = row >> 7, d = row & 127;
  float v = w[(size_t)row * KVL_ + c];
  if (d < NOPE_) nopeT[((size_t)h * KVL_ + c) * NOPE_ + d] = (bf16)v;
  else           vmat[((size_t)h * VD_ + (d - NOPE_)) * KVL_ + c] = (bf16)v;
}

// q rope: q f32 [tok,1536] -> qnope bf16 [b,h,s,64], qcat[...,256:288] = rope(q_pe)
__global__ __launch_bounds__(256)
void k_qrope(const float* __restrict__ q, bf16* __restrict__ qnope, bf16* __restrict__ qcat)
{
  int tkn = blockIdx.x;
  int b = tkn / S_, s = tkn % S_;
  const float* qr = q + (size_t)tkn * (H_ * (NOPE_ + ROPE_));
  for (int i = threadIdx.x; i < H_ * NOPE_; i += 256) {
    int h = i >> 6, d = i & 63;
    qnope[(((size_t)(b * H_ + h)) * S_ + s) * NOPE_ + d] = (bf16)qr[h * 96 + d];
  }
  int i = threadIdx.x;            // 256 threads == H_*16 rope pairs
  int h = i >> 4, fi = i & 15;
  float inv = powf(10000.0f, -((float)(2 * fi)) / 32.0f);
  float ang = (float)s * inv;
  float c = cosf(ang), sn = sinf(ang);
  float x1 = qr[h * 96 + NOPE_ + fi];
  float x2 = qr[h * 96 + NOPE_ + 16 + fi];
  size_t base = (((size_t)(b * H_ + h)) * S_ + s) * 288 + 256;
  qcat[base + fi]      = (bf16)(x1 * c - x2 * sn);
  qcat[base + 16 + fi] = (bf16)(x2 * c + x1 * sn);
}

// kv prep: kvp f32 [tok,288] -> kcat bf16 [b,s,288] (rmsnorm kv + rope k_pe)
//          and kvcT bf16 [b,256,1024]
__global__ __launch_bounds__(256)
void k_kvprep(const float* __restrict__ kvp, const float* __restrict__ kvw,
              bf16* __restrict__ kcat, bf16* __restrict__ kvcT)
{
  __shared__ float red[256];
  int tkn = blockIdx.x;
  int b = tkn / S_, s = tkn % S_;
  const float* xr = kvp + (size_t)tkn * (KVL_ + ROPE_);
  int t = threadIdx.x;
  float v = xr[t];                       // t < 256 covers kv part
  red[t] = v * v;
  __syncthreads();
  for (int sdx = 128; sdx > 0; sdx >>= 1) {
    if (t < sdx) red[t] += red[t + sdx];
    __syncthreads();
  }
  float r = rsqrtf(red[0] / (float)KVL_ + 1e-6f);
  float kn = kvw[t] * v * r;
  kcat[(size_t)tkn * 288 + t] = (bf16)kn;
  kvcT[((size_t)b * KVL_ + t) * S_ + s] = (bf16)kn;
  if (t < 16) {
    float inv = powf(10000.0f, -((float)(2 * t)) / 32.0f);
    float ang = (float)s * inv;
    float c = cosf(ang), sn = sinf(ang);
    float x1 = xr[KVL_ + t], x2 = xr[KVL_ + 16 + t];
    kcat[(size_t)tkn * 288 + 256 + t]      = (bf16)(x1 * c - x2 * sn);
    kcat[(size_t)tkn * 288 + 256 + 16 + t] = (bf16)(x2 * c + x1 * sn);
  }
}

// causal softmax over one score row (bf16 in-place), matching reference masking
__global__ __launch_bounds__(256)
void k_softmax(bf16* __restrict__ sc, const int* __restrict__ amask)
{
  __shared__ float red[256];
  int z = blockIdx.y;               // b*H + h
  int s = blockIdx.x;
  int b = z >> 4;
  bf16* row = sc + ((size_t)z * S_ + s) * S_;
  int t = threadIdx.x;

  float lv[4];
  float m = -3.0e38f;
#pragma unroll
  for (int i = 0; i < 4; ++i) {
    int tt = t + i * 256;
    float v = (float)row[tt];
    if (tt > s) v += NEGF;
    if (amask[b * S_ + tt] == 0) v += NEGF;
    lv[i] = v;
    m = fmaxf(m, v);
  }
  red[t] = m; __syncthreads();
  for (int sd = 128; sd > 0; sd >>= 1) {
    if (t < sd) red[t] = fmaxf(red[t], red[t + sd]);
    __syncthreads();
  }
  m = red[0];
  __syncthreads();

  float ev[4]; float sum = 0.0f;
#pragma unroll
  for (int i = 0; i < 4; ++i) { ev[i] = expf(lv[i] - m); sum += ev[i]; }
  red[t] = sum; __syncthreads();
  for (int sd = 128; sd > 0; sd >>= 1) {
    if (t < sd) red[t] += red[t + sd];
    __syncthreads();
  }
  float inv = 1.0f / (red[0] + 1e-30f);
#pragma unroll
  for (int i = 0; i < 4; ++i) {
    int tt = t + i * 256;
    row[tt] = (bf16)(ev[i] * inv);
  }
}

// gating: logits (written to output), top-2 routing, per-expert row lists
__global__ __launch_bounds__(64)
void k_gate(const bf16* __restrict__ y, const float* __restrict__ gw,
            const float* __restrict__ gb, float* __restrict__ logitsOut,
            int* __restrict__ counts, int* __restrict__ rowsIdx,
            float* __restrict__ wts)
{
  __shared__ float lg[E_];
  int tkn = blockIdx.x;
  int t = threadIdx.x;
  if (t < E_) {
    float sacc = gb[t];
    const bf16* yr = y + (size_t)tkn * D_;
    const float* wr = gw + (size_t)t * D_;
    for (int d = 0; d < D_; ++d) sacc += (float)yr[d] * wr[d];
    lg[t] = sacc;
    logitsOut[(size_t)tkn * E_ + t] = sacc;
  }
  __syncthreads();
  if (t == 0) {
    int e1 = 0; float b1v = lg[0];
    for (int e = 1; e < E_; ++e) if (lg[e] > b1v) { b1v = lg[e]; e1 = e; }
    int e2 = (e1 == 0) ? 1 : 0; float b2v = lg[e2];
    for (int e = 0; e < E_; ++e) if (e != e1 && lg[e] > b2v) { b2v = lg[e]; e2 = e; }
    float mm = fmaxf(b1v, b2v);
    float x1 = expf(b1v - mm), x2 = expf(b2v - mm);
    float inv = 1.0f / (x1 + x2);
    int s1 = atomicAdd(&counts[e1], 1);
    rowsIdx[e1 * TOK_ + s1] = tkn; wts[e1 * TOK_ + s1] = x1 * inv;
    int s2 = atomicAdd(&counts[e2], 1);
    rowsIdx[e2 * TOK_ + s2] = tkn; wts[e2 * TOK_ + s2] = x2 * inv;
  }
}

// ---------------------------------------------------------------------------
// Host-side pipeline
// ---------------------------------------------------------------------------
static inline size_t al256(size_t x) { return (x + 255) & ~(size_t)255; }

extern "C" void kernel_launch(void* const* d_in, const int* in_sizes, int n_in,
                              void* d_out, int out_size, void* d_ws, size_t ws_size,
                              hipStream_t stream) {
  const float* hidden = (const float*)d_in[0];
  const int*   amask  = (const int*)  d_in[1];
  const float* in_ln  = (const float*)d_in[2];
  const float* postln = (const float*)d_in[3];
  const float* wqa    = (const float*)d_in[4];
  const float* wqab   = (const float*)d_in[5];
  const float* qnw    = (const float*)d_in[6];
  const float* wqb    = (const float*)d_in[7];
  const float* wqbb   = (const float*)d_in[8];
  const float* wkva   = (const float*)d_in[9];
  const float* wkvab  = (const float*)d_in[10];
  const float* kvnw   = (const float*)d_in[11];
  const float* wkvb   = (const float*)d_in[12];
  const float* wo     = (const float*)d_in[13];
  const float* wob    = (const float*)d_in[14];
  const float* gw     = (const float*)d_in[15];
  const float* gb     = (const float*)d_in[16];
  const float* gproj  = (const float*)d_in[17];
  const float* uproj  = (const float*)d_in[18];
  const float* dproj  = (const float*)d_in[19];

  float* outF   = (float*)d_out;
  float* logits = outF + (size_t)TOK_ * D_;

  // ---- workspace layout ----
  char* wsb = (char*)d_ws;
  size_t off = 0;
  auto alloc = [&](size_t bytes) -> void* { void* p = wsb + off; off = al256(off + bytes); return p; };

  bf16* xbf    = (bf16*)alloc((size_t)TOK_ * D_ * 2);
  bf16* wqa_b  = (bf16*)alloc((size_t)QL_ * D_ * 2);
  bf16* wqb_b  = (bf16*)alloc((size_t)H_ * 96 * QL_ * 2);
  bf16* wkva_b = (bf16*)alloc((size_t)(KVL_ + ROPE_) * D_ * 2);
  bf16* wnT    = (bf16*)alloc((size_t)H_ * KVL_ * NOPE_ * 2);
  bf16* wvm    = (bf16*)alloc((size_t)H_ * VD_ * KVL_ * 2);
  bf16* wo_b   = (bf16*)alloc((size_t)D_ * D_ * 2);
  bf16* gp_b   = (bf16*)alloc((size_t)E_ * FE_ * D_ * 2);
  bf16* up_b   = (bf16*)alloc((size_t)E_ * FE_ * D_ * 2);
  bf16* dp_b   = (bf16*)alloc((size_t)E_ * D_ * FE_ * 2);
  float* qaf   = (float*)alloc((size_t)TOK_ * QL_ * 4);
  bf16* qanb   = (bf16*)alloc((size_t)TOK_ * QL_ * 2);
  float* qf    = (float*)alloc((size_t)TOK_ * H_ * 96 * 4);
  bf16* qnope  = (bf16*)alloc((size_t)B_ * H_ * S_ * NOPE_ * 2);
  bf16* qcat   = (bf16*)alloc((size_t)B_ * H_ * S_ * 288 * 2);
  float* kvpf  = (float*)alloc((size_t)TOK_ * 288 * 4);
  bf16* kcat   = (bf16*)alloc((size_t)TOK_ * 288 * 2);
  bf16* kvcT   = (bf16*)alloc((size_t)B_ * KVL_ * S_ * 2);
  bf16* sc     = (bf16*)alloc((size_t)B_ * H_ * S_ * S_ * 2);
  bf16* octx   = (bf16*)alloc((size_t)B_ * S_ * H_ * KVL_ * 2);
  bf16* attnc  = (bf16*)alloc((size_t)TOK_ * D_ * 2);
  float* hf    = (float*)alloc((size_t)TOK_ * D_ * 4);
  bf16* ybf    = (bf16*)alloc((size_t)TOK_ * D_ * 2);
  int*  counts = (int*) alloc(E_ * 4);
  int*  rowsI  = (int*) alloc((size_t)E_ * TOK_ * 4);
  float* wtsP  = (float*)alloc((size_t)E_ * TOK_ * 4);
  bf16* hact   = (bf16*)alloc((size_t)E_ * TOK_ * FE_ * 2);
  (void)ws_size;

  // ---- weight conversion to bf16 ----
  k_zero_i32<<<1, 32, 0, stream>>>(counts, E_);
  k_cast_bf16<<<2048, 256, 0, stream>>>(wqa,   wqa_b,  (size_t)QL_ * D_);
  k_cast_bf16<<<2048, 256, 0, stream>>>(wqb,   wqb_b,  (size_t)H_ * 96 * QL_);
  k_cast_bf16<<<2048, 256, 0, stream>>>(wkva,  wkva_b, (size_t)(KVL_ + ROPE_) * D_);
  k_cast_bf16<<<2048, 256, 0, stream>>>(wo,    wo_b,   (size_t)D_ * D_);
  k_cast_bf16<<<8192, 256, 0, stream>>>(gproj, gp_b,   (size_t)E_ * FE_ * D_);
  k_cast_bf16<<<8192, 256, 0, stream>>>(uproj, up_b,   (size_t)E_ * FE_ * D_);
  k_cast_bf16<<<8192, 256, 0, stream>>>(dproj, dp_b,   (size_t)E_ * D_ * FE_);
  k_wkvb_split<<<2048, 256, 0, stream>>>(wkvb, wnT, wvm);

  // ---- attention path ----
  k_rmsnorm<<<TOK_, 256, 0, stream>>>(hidden, in_ln, xbf, D_);

  // q_a = x @ wq_a^T + b
  k_gemm<false><<<dim3(QL_ / 64, TOK_ / 64, 1), 128, 0, stream>>>(
      xbf, wqa_b, wqab, nullptr, qaf, TOK_, QL_, D_, D_, D_, QL_,
      0, 0, 0, 0, 0, 0, 1.0f, 0, 0, 1);
  k_rmsnorm<<<TOK_, 256, 0, stream>>>(qaf, qnw, qanb, QL_);

  // q = q_a @ wq_b^T + b
  k_gemm<false><<<dim3(H_ * 96 / 64, TOK_ / 64, 1), 128, 0, stream>>>(
      qanb, wqb_b, wqbb, nullptr, qf, TOK_, H_ * 96, QL_, QL_, QL_, H_ * 96,
      0, 0, 0, 0, 0, 0, 1.0f, 0, 0, 1);
  k_qrope<<<TOK_, 256, 0, stream>>>(qf, qnope, qcat);

  // kvp = x @ wkv_a^T + b
  k_gemm<false><<<dim3((KVL_ + ROPE_ + 63) / 64, TOK_ / 64, 1), 128, 0, stream>>>(
      xbf, wkva_b, wkvab, nullptr, kvpf, TOK_, KVL_ + ROPE_, D_, D_, D_, KVL_ + ROPE_,
      0, 0, 0, 0, 0, 0, 1.0f, 0, 0, 1);
  k_kvprep<<<TOK_, 256, 0, stream>>>(kvpf, kvnw, kcat, kvcT);

  // q_abs[b,h] = q_nope[b,h] @ wkv_b_nope[h]   -> qcat[...,0:256]
  k_gemm<true><<<dim3(KVL_ / 64, S_ / 64, B_ * H_), 128, 0, stream>>>(
      qnope, wnT, nullptr, nullptr, qcat, S_, KVL_, NOPE_, NOPE_, NOPE_, 288,
      (long long)H_ * S_ * NOPE_, (long long)S_ * NOPE_,
      0, (long long)KVL_ * NOPE_,
      (long long)H_ * S_ * 288, (long long)S_ * 288, 1.0f, 0, 0, H_);

  // scores[b,h] = qcat[b,h] @ kcat[b]^T / sqrt(96)   (causal tile skip)
  k_gemm<true><<<dim3(S_ / 64, S_ / 64, B_ * H_), 128, 0, stream>>>(
      qcat, kcat, nullptr, nullptr, sc, S_, S_, 288, 288, 288, S_,
      (long long)H_ * S_ * 288, (long long)S_ * 288,
      (long long)S_ * 288, 0,
      (long long)H_ * S_ * S_, (long long)S_ * S_,
      0.10206207261596575f /* 1/sqrt(96) */, 1, 0, H_);

  k_softmax<<<dim3(S_, B_ * H_), 256, 0, stream>>>(sc, amask);

  // o[b,h] = probs @ kv_c[b]   (K limited causally)
  k_gemm<true><<<dim3(KVL_ / 64, S_ / 64, B_ * H_), 128, 0, stream>>>(
      sc, kvcT, nullptr, nullptr, octx, S_, KVL_, S_, S_, S_, H_ * KVL_,
      (long long)H_ * S_ * S_, (long long)S_ * S_,
      (long long)KVL_ * S_, 0,
      (long long)S_ * H_ * KVL_, (long long)KVL_, 1.0f, 0, 1, H_);

  // o_proj[b,h] = o[b,h] @ wkv_b_v[h]^T  -> attncat[b,s,h*64]
  k_gemm<true><<<dim3(1, S_ / 64, B_ * H_), 128, 0, stream>>>(
      octx, wvm, nullptr, nullptr, attnc, S_, VD_, KVL_, H_ * KVL_, KVL_, H_ * VD_,
      (long long)S_ * H_ * KVL_, (long long)KVL_,
      0, (long long)VD_ * KVL_,
      (long long)S_ * H_ * VD_, (long long)VD_, 1.0f, 0, 0, H_);

  // h = attn @ wo^T + wo_b + hidden
  k_gemm<false><<<dim3(D_ / 64, TOK_ / 64, 1), 128, 0, stream>>>(
      attnc, wo_b, wob, hidden, hf, TOK_, D_, D_, D_, D_, D_,
      0, 0, 0, 0, 0, 0, 1.0f, 0, 0, 1);

  // ---- MoE path ----
  k_rmsnorm<<<TOK_, 256, 0, stream>>>(hf, postln, ybf, D_);
  k_copy_f32<<<4096, 256, 0, stream>>>(hf, outF, (size_t)TOK_ * D_);   // out = h
  k_gate<<<TOK_, 64, 0, stream>>>(ybf, gw, gb, logits, counts, rowsI, wtsP);
  k_moe_gateup<<<dim3(FE_ / 64, TOK_ / 64, E_), 128, 0, stream>>>(
      ybf, gp_b, up_b, counts, rowsI, hact);
  k_moe_down<<<dim3(D_ / 64, TOK_ / 64, E_), 128, 0, stream>>>(
      hact, dp_b, counts, rowsI, wtsP, outF);
}
